// BitNetGQA_73443940762132
// MI455X (gfx1250) — 16-bit and 8-bit matrix operand order repaired and checked by reading, compiled, not run on hardware
//
#include <hip/hip_runtime.h>
#include <cstdint>
#include <cstddef>

typedef __attribute__((ext_vector_type(4)))  int      v4i;
typedef __attribute__((ext_vector_type(8)))  int      v8i;
typedef __attribute__((ext_vector_type(8)))  float    v8f;
typedef __attribute__((ext_vector_type(16))) _Float16 v16h;

typedef __attribute__((address_space(1))) v4i* gv4p;
typedef __attribute__((address_space(3))) v4i* lv4p;

namespace {
constexpr int S   = 2048;
constexpr int H   = 4096;
constexpr int KVD = 1024;
constexpr int D   = 128;
constexpr int NH  = 32;
constexpr int G   = 8;
constexpr int LDA = 80;

union V16 { v16h h; int i[8]; };
}

#if defined(__has_builtin)
#if __has_builtin(__builtin_amdgcn_global_load_async_to_lds_b128)
#define HAVE_ASYNC_LDS 1
#endif
#endif

__device__ __forceinline__ void copy16(void* l, const void* g) {
#ifdef HAVE_ASYNC_LDS
  __builtin_amdgcn_global_load_async_to_lds_b128(
      (gv4p)(uintptr_t)g,
      (lv4p)(uint32_t)(uintptr_t)l,
      0, 0);
#else
  *(int4*)l = *(const int4*)g;
#endif
}

template <int N>
__device__ __forceinline__ void wait_async() {
#ifdef HAVE_ASYNC_LDS
#if __has_builtin(__builtin_amdgcn_s_wait_asynccnt)
  __builtin_amdgcn_s_wait_asynccnt(N);
#else
  asm volatile("s_wait_asynccnt %0" ::"i"(N) : "memory");
#endif
#endif
}

__global__ __launch_bounds__(256) void k_wabs_partial(
    const float* __restrict__ w0, const float* __restrict__ w1,
    const float* __restrict__ w2, const float* __restrict__ w3,
    float* __restrict__ partials) {
  const float* wp[4] = {w0, w1, w2, w3};
  const size_t ns[4] = {(size_t)H * H, (size_t)KVD * H, (size_t)KVD * H, (size_t)H * H};
  const int t = blockIdx.y;
  const float* w = wp[t];
  const size_t n = ns[t];
  float s = 0.f;
  for (size_t i = (size_t)blockIdx.x * 256 + threadIdx.x; i < n; i += (size_t)1024 * 256)
    s += fabsf(w[i]);
  __shared__ float red[256];
  red[threadIdx.x] = s;
  __syncthreads();
  for (int st = 128; st > 0; st >>= 1) {
    if ((int)threadIdx.x < st) red[threadIdx.x] += red[threadIdx.x + st];
    __syncthreads();
  }
  if (threadIdx.x == 0) partials[t * 1024 + blockIdx.x] = red[0];
}

__global__ __launch_bounds__(256) void k_wabs_final(const float* __restrict__ partials,
                                                    float* __restrict__ wdeq) {
  const size_t ns[4] = {(size_t)H * H, (size_t)KVD * H, (size_t)KVD * H, (size_t)H * H};
  const int t = blockIdx.x;
  float s = 0.f;
  for (int i = threadIdx.x; i < 1024; i += 256) s += partials[t * 1024 + i];
  __shared__ float red[256];
  red[threadIdx.x] = s;
  __syncthreads();
  for (int st = 128; st > 0; st >>= 1) {
    if ((int)threadIdx.x < st) red[threadIdx.x] += red[threadIdx.x + st];
    __syncthreads();
  }
  if (threadIdx.x == 0) {
    float mean = red[0] / (float)ns[t];
    wdeq[t] = fmaxf(mean, 1e-5f);
  }
}

__global__ __launch_bounds__(256) void k_wquant(const float* __restrict__ w,
                                                signed char* __restrict__ q,
                                                const float* __restrict__ wdeq,
                                                int t, size_t n) {
  size_t i = (size_t)blockIdx.x * 256 + threadIdx.x;
  if (i >= n) return;
  float scale = 1.f / wdeq[t];
  float v = fminf(fmaxf(rintf(w[i] * scale), -1.f), 1.f);
  q[i] = (signed char)v;
}

__global__ __launch_bounds__(256) void k_ln_quant(
    const float* __restrict__ x, const float* __restrict__ gw,
    const float* __restrict__ gb, signed char* __restrict__ q,
    float* __restrict__ adeq) {
  __shared__ float red[256];
  const int tkn = blockIdx.x, tid = threadIdx.x;
  const float* row = x + (size_t)tkn * H;
  float r[16];
  float s = 0.f;
#pragma unroll
  for (int i = 0; i < 16; ++i) { r[i] = row[tid + 256 * i]; s += r[i]; }
  red[tid] = s;
  __syncthreads();
  for (int st = 128; st > 0; st >>= 1) {
    if (tid < st) red[tid] += red[tid + st];
    __syncthreads();
  }
  float mu = red[0] * (1.f / (float)H);
  __syncthreads();
  s = 0.f;
#pragma unroll
  for (int i = 0; i < 16; ++i) { float d = r[i] - mu; s += d * d; }
  red[tid] = s;
  __syncthreads();
  for (int st = 128; st > 0; st >>= 1) {
    if (tid < st) red[tid] += red[tid + st];
    __syncthreads();
  }
  float inv = rsqrtf(red[0] * (1.f / (float)H) + 1e-5f);
  __syncthreads();
  float am = 0.f;
#pragma unroll
  for (int i = 0; i < 16; ++i) {
    r[i] = (r[i] - mu) * inv * gw[tid + 256 * i] + gb[tid + 256 * i];
    am = fmaxf(am, fabsf(r[i]));
  }
  red[tid] = am;
  __syncthreads();
  for (int st = 128; st > 0; st >>= 1) {
    if (tid < st) red[tid] = fmaxf(red[tid], red[tid + st]);
    __syncthreads();
  }
  float amax = fmaxf(red[0], 1e-5f);
  float scale = 127.f / amax;
#pragma unroll
  for (int i = 0; i < 16; ++i) {
    float qq = fminf(fmaxf(rintf(r[i] * scale), -128.f), 127.f);
    q[(size_t)tkn * H + tid + 256 * i] = (signed char)qq;
  }
  if (tid == 0) adeq[tkn] = amax * (1.f / 127.f);
}

__global__ __launch_bounds__(256) void k_actquant(const float* __restrict__ x,
                                                  signed char* __restrict__ q,
                                                  float* __restrict__ adeq) {
  __shared__ float red[256];
  const int tkn = blockIdx.x, tid = threadIdx.x;
  const float* row = x + (size_t)tkn * H;
  float r[16];
  float am = 0.f;
#pragma unroll
  for (int i = 0; i < 16; ++i) { r[i] = row[tid + 256 * i]; am = fmaxf(am, fabsf(r[i])); }
  red[tid] = am;
  __syncthreads();
  for (int st = 128; st > 0; st >>= 1) {
    if (tid < st) red[tid] = fmaxf(red[tid], red[tid + st]);
    __syncthreads();
  }
  float amax = fmaxf(red[0], 1e-5f);
  float scale = 127.f / amax;
#pragma unroll
  for (int i = 0; i < 16; ++i) {
    float qq = fminf(fmaxf(rintf(r[i] * scale), -128.f), 127.f);
    q[(size_t)tkn * H + tid + 256 * i] = (signed char)qq;
  }
  if (tid == 0) adeq[tkn] = amax * (1.f / 127.f);
}

__device__ __forceinline__ void stage_tile_i8(signed char* Asb, signed char* Bsb,
                                              const signed char* __restrict__ A,
                                              const signed char* __restrict__ Bw,
                                              int m0, int n0, int k0, int tid) {
#pragma unroll
  for (int rep = 0; rep < 2; ++rep) {
    int idx = tid + rep * 256;
    int row = idx >> 2, piece = idx & 3;
    copy16(Asb + row * LDA + piece * 16, A + (size_t)(m0 + row) * H + k0 + piece * 16);
    copy16(Bsb + row * LDA + piece * 16, Bw + (size_t)(n0 + row) * H + k0 + piece * 16);
  }
}

__global__ __launch_bounds__(256) void k_gemm_i8(
    const signed char* __restrict__ A, const signed char* __restrict__ Bw,
    const float* __restrict__ adeq, const float* __restrict__ wdeq, int wi,
    float* __restrict__ Y, int N) {
  constexpr int NT = H / 64;
  __shared__ signed char As[2][128 * LDA];
  __shared__ signed char Bs[2][128 * LDA];

  const int m0 = blockIdx.y * 128, n0 = blockIdx.x * 128;
  const int tid = threadIdx.x;
  const int wave = tid >> 5, lane = tid & 31;
  const int half = lane >> 4, li = lane & 15;
  const int mbase = (wave & 3) * 32;
  const int nbase = (wave >> 2) * 64;

  v8i acc[2][4];
#pragma unroll
  for (int i = 0; i < 2; ++i)
#pragma unroll
    for (int j = 0; j < 4; ++j) acc[i][j] = (v8i){0, 0, 0, 0, 0, 0, 0, 0};

  stage_tile_i8(As[0], Bs[0], A, Bw, m0, n0, 0, tid);

  for (int kt = 0; kt < NT; ++kt) {
    const int cur = kt & 1;
    __syncthreads();
    if (kt + 1 < NT) {
      stage_tile_i8(As[1 - cur], Bs[1 - cur], A, Bw, m0, n0, (kt + 1) * 64, tid);
      wait_async<4>();
    } else {
      wait_async<0>();
    }
    __syncthreads();

    v8i af[2], bf[4];
#pragma unroll
    for (int i = 0; i < 2; ++i) {
      const int row = mbase + 16 * i + li;
#pragma unroll
      for (int v = 0; v < 8; ++v) {
        const int kb = 32 * (v >> 2) + 16 * ((v >> 1) & 1) + 4 * (v & 1) + 8 * half;
        af[i][v] = *(const int*)(As[cur] + row * LDA + kb);
      }
    }
#pragma unroll
    for (int j = 0; j < 4; ++j) {
      const int col = nbase + 16 * j + li;
#pragma unroll
      for (int v = 0; v < 8; ++v) {
        const int kb = 32 * (v >> 2) + 16 * ((v >> 1) & 1) + 4 * (v & 1) + 8 * half;
        bf[j][v] = *(const int*)(Bs[cur] + col * LDA + kb);
      }
    }
#pragma unroll
    for (int i = 0; i < 2; ++i)
#pragma unroll
      for (int j = 0; j < 4; ++j)
        acc[i][j] = __builtin_amdgcn_wmma_i32_16x16x64_iu8(
            true, af[i], true, bf[j], acc[i][j], false, false);
  }

  const float wq = wdeq[wi];
#pragma unroll
  for (int i = 0; i < 2; ++i)
#pragma unroll
    for (int j = 0; j < 4; ++j) {
#pragma unroll
      for (int v = 0; v < 8; ++v) {
        const int m = m0 + mbase + 16 * i + v + 8 * half;
        const int n = n0 + nbase + 16 * j + li;
        Y[(size_t)m * N + n] = (float)acc[i][j][v] * adeq[m] * wq;
      }
    }
}

__global__ __launch_bounds__(256) void k_rope(const float* __restrict__ X,
                                              _Float16* __restrict__ out,
                                              int heads, float mul) {
  const size_t idx = (size_t)blockIdx.x * 256 + threadIdx.x;
  const int i = (int)(idx & 63);
  const size_t rest = idx >> 6;
  const int h = (int)(rest % heads);
  const int t = (int)(rest / heads);
  if (t >= S) return;
  const float invf = exp2f(-(float)i * 0.20762050593043f);
  const float ang = (float)t * invf;
  const float c = cosf(ang), sn = sinf(ang);
  const float x1 = X[((size_t)t * heads + h) * D + i];
  const float x2 = X[((size_t)t * heads + h) * D + i + 64];
  out[((size_t)h * S + t) * D + i]      = (_Float16)((x1 * c - x2 * sn) * mul);
  out[((size_t)h * S + t) * D + i + 64] = (_Float16)((x2 * c + x1 * sn) * mul);
}

__global__ __launch_bounds__(256) void k_tof16(const float* __restrict__ X,
                                               _Float16* __restrict__ out) {
  const size_t idx = (size_t)blockIdx.x * 256 + threadIdx.x;
  const int t = (int)(idx / KVD);
  const int c = (int)(idx % KVD);
  const int h = c >> 7, d = c & 127;
  out[((size_t)h * S + t) * D + d] = (_Float16)X[idx];
}

__global__ __launch_bounds__(128) void k_attn(const _Float16* __restrict__ qh,
                                              const _Float16* __restrict__ kh,
                                              const _Float16* __restrict__ vh,
                                              float* __restrict__ aout) {
  constexpr int LDQ = 136;
  constexpr int LDP = 72;
  __shared__ _Float16 Qs[64 * LDQ];
  __shared__ _Float16 Ks[64 * LDQ];
  __shared__ _Float16 Vs[64 * LDQ];
  __shared__ _Float16 Pt[4 * 16 * LDP];

  const int qt = blockIdx.x, h = blockIdx.y;
  const int g = h >> 2;
  const int tid = threadIdx.x;
  const int w = tid >> 5, lane = tid & 31;
  const int half = lane >> 4, li = lane & 15;

#pragma unroll
  for (int it = 0; it < 8; ++it) {
    int idx = tid + it * 128;
    int row = idx >> 4, piece = idx & 15;
    copy16(Qs + row * LDQ + piece * 8,
           qh + ((size_t)h * S + qt * 64 + row) * D + piece * 8);
  }
  wait_async<0>();
  __syncthreads();

  V16 qa[4];
  const int qrow = w * 16 + li;
#pragma unroll
  for (int c = 0; c < 4; ++c)
#pragma unroll
    for (int v = 0; v < 8; ++v) {
      const int kb = c * 32 + 16 * (v >> 2) + 2 * (v & 3) + 8 * half;
      qa[c].i[v] = *(const int*)(Qs + qrow * LDQ + kb);
    }

  v8f of[8];
#pragma unroll
  for (int dj = 0; dj < 8; ++dj) of[dj] = (v8f){0, 0, 0, 0, 0, 0, 0, 0};
  float mrow[8], lsum[8];
#pragma unroll
  for (int v = 0; v < 8; ++v) { mrow[v] = -1e30f; lsum[v] = 0.f; }

  for (int kc = 0; kc < S / 64; ++kc) {
    __syncthreads();
#pragma unroll
    for (int it = 0; it < 8; ++it) {
      int idx = tid + it * 128;
      int row = idx >> 4, piece = idx & 15;
      copy16(Ks + row * LDQ + piece * 8,
             kh + ((size_t)g * S + kc * 64 + row) * D + piece * 8);
      copy16(Vs + row * LDQ + piece * 8,
             vh + ((size_t)g * S + kc * 64 + row) * D + piece * 8);
    }
    wait_async<0>();
    __syncthreads();

    v8f sf[4];
#pragma unroll
    for (int j = 0; j < 4; ++j) {
      sf[j] = (v8f){0, 0, 0, 0, 0, 0, 0, 0};
#pragma unroll
      for (int c = 0; c < 4; ++c) {
        V16 bf;
#pragma unroll
        for (int v = 0; v < 8; ++v) {
          const int kb = c * 32 + 16 * (v >> 2) + 2 * (v & 3) + 8 * half;
          bf.i[v] = *(const int*)(Ks + (j * 16 + li) * LDQ + kb);
        }
        sf[j] = __builtin_amdgcn_wmma_f32_16x16x32_f16(
            false, qa[c].h, false, bf.h, (short)0, sf[j], false, false);
      }
    }

    float corr[8];
#pragma unroll
    for (int v = 0; v < 8; ++v) {
      float mx = fmaxf(fmaxf(sf[0][v], sf[1][v]), fmaxf(sf[2][v], sf[3][v]));
      mx = fmaxf(mx, __shfl_xor(mx, 8, 32));
      mx = fmaxf(mx, __shfl_xor(mx, 4, 32));
      mx = fmaxf(mx, __shfl_xor(mx, 2, 32));
      mx = fmaxf(mx, __shfl_xor(mx, 1, 32));
      const float nm = fmaxf(mrow[v], mx);
      const float cr = __expf(mrow[v] - nm);
      float ps = 0.f;
#pragma unroll
      for (int j = 0; j < 4; ++j) {
        float p = __expf(sf[j][v] - nm);
        sf[j][v] = p;
        ps += p;
      }
      ps += __shfl_xor(ps, 8, 32);
      ps += __shfl_xor(ps, 4, 32);
      ps += __shfl_xor(ps, 2, 32);
      ps += __shfl_xor(ps, 1, 32);
      lsum[v] = lsum[v] * cr + ps;
      mrow[v] = nm;
      corr[v] = cr;
    }
#pragma unroll
    for (int dj = 0; dj < 8; ++dj)
#pragma unroll
      for (int v = 0; v < 8; ++v) of[dj][v] *= corr[v];

#pragma unroll
    for (int j = 0; j < 4; ++j)
#pragma unroll
      for (int v = 0; v < 8; ++v)
        Pt[(w * 16 + v + 8 * half) * LDP + j * 16 + li] = (_Float16)sf[j][v];
    __syncthreads();

#pragma unroll
    for (int c2 = 0; c2 < 2; ++c2) {
      V16 pa;
#pragma unroll
      for (int v = 0; v < 8; ++v) {
        const int kb = 16 * (v >> 2) + 2 * (v & 3) + 8 * half;
        pa.i[v] = *(const int*)(Pt + (w * 16 + li) * LDP + c2 * 32 + kb);
      }
#pragma unroll
      for (int dj = 0; dj < 8; ++dj) {
        V16 vf;
#pragma unroll
        for (int v = 0; v < 8; ++v) {
          const int k = c2 * 32 + 16 * (v >> 2) + 2 * (v & 3) + 8 * half;
          unsigned lo = *(const unsigned short*)(Vs + k * LDQ + dj * 16 + li);
          unsigned hi = *(const unsigned short*)(Vs + (k + 1) * LDQ + dj * 16 + li);
          vf.i[v] = (int)(lo | (hi << 16));
        }
        of[dj] = __builtin_amdgcn_wmma_f32_16x16x32_f16(
            false, pa.h, false, vf.h, (short)0, of[dj], false, false);
      }
    }
  }

#pragma unroll
  for (int v = 0; v < 8; ++v) {
    const float inv = 1.f / lsum[v];
    const int tkn = qt * 64 + w * 16 + v + 8 * half;
#pragma unroll
    for (int dj = 0; dj < 8; ++dj)
      aout[(size_t)tkn * H + h * D + dj * 16 + li] = of[dj][v] * inv;
  }
}

extern "C" void kernel_launch(void* const* d_in, const int* in_sizes, int n_in,
                              void* d_out, int out_size, void* d_ws, size_t ws_size,
                              hipStream_t stream) {
  (void)in_sizes; (void)n_in; (void)out_size; (void)ws_size;
  const float* hs  = (const float*)d_in[0];
  const float* lnw = (const float*)d_in[1];
  const float* lnb = (const float*)d_in[2];
  const float* Wq  = (const float*)d_in[3];
  const float* Wk  = (const float*)d_in[4];
  const float* Wv  = (const float*)d_in[5];
  const float* Wo  = (const float*)d_in[6];

  char* base = (char*)d_ws;
  size_t off = 0;
  auto take = [&](size_t bytes) -> char* {
    char* p = base + off;
    off = (off + bytes + 255) & ~(size_t)255;
    return p;
  };

  float* partials = (float*)take(4096 * sizeof(float));
  float* wdeq     = (float*)take(4 * sizeof(float));
  signed char* wq8 = (signed char*)take((size_t)H * H);
  signed char* wk8 = (signed char*)take((size_t)KVD * H);
  signed char* wv8 = (signed char*)take((size_t)KVD * H);
  signed char* wo8 = (signed char*)take((size_t)H * H);
  signed char* act8 = (signed char*)take((size_t)S * H);
  float* adeq  = (float*)take((size_t)S * sizeof(float));
  float* qf32  = (float*)take((size_t)S * H * sizeof(float));
  float* kf32  = (float*)take((size_t)S * KVD * sizeof(float));
  float* vf32  = (float*)take((size_t)S * KVD * sizeof(float));
  _Float16* qhh = (_Float16*)take((size_t)NH * S * D * sizeof(_Float16));
  _Float16* khh = (_Float16*)take((size_t)G * S * D * sizeof(_Float16));
  _Float16* vhh = (_Float16*)take((size_t)G * S * D * sizeof(_Float16));
  float* attnout = qf32;

  k_wabs_partial<<<dim3(1024, 4), 256, 0, stream>>>(Wq, Wk, Wv, Wo, partials);
  k_wabs_final<<<4, 256, 0, stream>>>(partials, wdeq);
  k_wquant<<<(int)(((size_t)H * H + 255) / 256), 256, 0, stream>>>(Wq, wq8, wdeq, 0, (size_t)H * H);
  k_wquant<<<(int)(((size_t)KVD * H + 255) / 256), 256, 0, stream>>>(Wk, wk8, wdeq, 1, (size_t)KVD * H);
  k_wquant<<<(int)(((size_t)KVD * H + 255) / 256), 256, 0, stream>>>(Wv, wv8, wdeq, 2, (size_t)KVD * H);
  k_wquant<<<(int)(((size_t)H * H + 255) / 256), 256, 0, stream>>>(Wo, wo8, wdeq, 3, (size_t)H * H);

  k_ln_quant<<<S, 256, 0, stream>>>(hs, lnw, lnb, act8, adeq);

  k_gemm_i8<<<dim3(H / 128, S / 128), 256, 0, stream>>>(act8, wq8, adeq, wdeq, 0, qf32, H);
  k_gemm_i8<<<dim3(KVD / 128, S / 128), 256, 0, stream>>>(act8, wk8, adeq, wdeq, 1, kf32, KVD);
  k_gemm_i8<<<dim3(KVD / 128, S / 128), 256, 0, stream>>>(act8, wv8, adeq, wdeq, 2, vf32, KVD);

  k_rope<<<(int)(((size_t)S * NH * 64) / 256), 256, 0, stream>>>(qf32, qhh, NH, 0.08838834764831845f);
  k_rope<<<(int)(((size_t)S * G * 64) / 256), 256, 0, stream>>>(kf32, khh, G, 1.0f);
  k_tof16<<<(int)(((size_t)S * KVD) / 256), 256, 0, stream>>>(vf32, vhh);

  k_attn<<<dim3(S / 64, NH), 128, 0, stream>>>(qhh, khh, vhh, attnout);

  k_actquant<<<S, 256, 0, stream>>>(attnout, act8, adeq);
  k_gemm_i8<<<dim3(H / 128, S / 128), 256, 0, stream>>>(act8, wo8, adeq, wdeq, 3, (float*)d_out, H);
}
